// RNN_27255862460472
// MI455X (gfx1250) — compile-verified, matrix operand order checked by reading, not run on hardware
//
#include <hip/hip_runtime.h>
#include <hip/hip_bf16.h>
#include <math.h>

// ---------------------------------------------------------------------------
// Vanilla tanh RNN:  h_t = tanh(x_t @ Wx + b + h_{t-1} @ Wh),  out = all h_t
// N=128, T=256, D=H=1024.
// Phase 1: xW (32768x1024 GEMM, bf16 WMMA, f32 accum) -> d_out.
//          Double-buffered LDS, distance-2 pipeline, async global->LDS for B.
//          Steady-state loop is branch-free; last two chunks peeled for drain.
// Phase 2: ONE persistent kernel, 64 blocks, Wh slice LDS-resident, 256 steps
//          separated by an agent-scope atomic grid barrier.
// ---------------------------------------------------------------------------

typedef __bf16 bf16_t;
typedef __attribute__((ext_vector_type(16))) __bf16 v16bf;
typedef __attribute__((ext_vector_type(8)))  __bf16 v8bf;
typedef __attribute__((ext_vector_type(8)))  float  v8f;

union Frag16 { v16bf v; v8bf h[2]; };

__device__ inline bf16_t f2bf(float f) { return (bf16_t)f; }  // v_cvt_pk_bf16_f32

// --- CDNA5 async global->LDS copy (ASYNCcnt-tracked) -----------------------
__device__ inline unsigned lds_off32(const void* p) {
  // generic LDS pointer: addr[31:0] is the LDS byte address (aperture rule)
  return (unsigned)(uintptr_t)p;
}
__device__ inline void async_copy_b128(unsigned lds_dst, const void* gsrc) {
  asm volatile("global_load_async_to_lds_b128 %0, %1, off"
               :: "v"(lds_dst), "v"((unsigned long long)(uintptr_t)gsrc)
               : "memory");
}
__device__ inline void wait_async0() {
  asm volatile("s_wait_asynccnt 0x0" ::: "memory");
}
__device__ inline void wait_async2() {  // allow 2 copies (next chunk) in flight
  asm volatile("s_wait_asynccnt 0x2" ::: "memory");
}

// ---------------------- one-time weight conversion -------------------------

__global__ void cvt_transpose(const float* __restrict__ in, bf16_t* __restrict__ outT,
                              int R, int C) {
  int i = blockIdx.x * blockDim.x + threadIdx.x;
  if (i >= R * C) return;
  int r = i / C, c = i % C;
  outT[(size_t)c * R + r] = f2bf(in[i]);
}

__global__ void cvt(const float* __restrict__ in, bf16_t* __restrict__ out, int n) {
  int i = blockIdx.x * blockDim.x + threadIdx.x;
  if (i < n) out[i] = f2bf(in[i]);
}

__global__ void zero_u32(unsigned* p) {
  if (threadIdx.x == 0) *p = 0u;
}

// ---------------------- phase 1: xW = x @ Wx + b ---------------------------
// Block: 256 thr / 8 waves, tile 64(M) x 128(N); wave tile 32x32 (2x2 frags).

#define BM  64
#define BN  128
#define BK  32
#define NCH 32   // 1024 / BK
#define LDA 40   // padded LDS row stride (halves): 80B, 16B-aligned, bank-safe
#define LDB 40

__global__ __launch_bounds__(256) void xw_gemm(
    const float*  __restrict__ x,     // (32768, 1024)
    const bf16_t* __restrict__ WxT,   // (1024, 1024)  [n][k]
    const float*  __restrict__ bias,  // (1024)
    float*        __restrict__ out)   // (32768, 1024)
{
  __shared__ __align__(16) bf16_t As[2][BM * LDA];
  __shared__ __align__(16) bf16_t Bs[2][BN * LDB];

  const int K = 1024, NC = 1024;
  const int r0 = blockIdx.x * BM;
  const int n0 = blockIdx.y * BN;

  const int tid  = threadIdx.x;
  const int lane = tid & 31;
  const int wave = tid >> 5;
  const int wm   = wave >> 2;      // 0..1
  const int wn   = wave & 3;       // 0..3
  const int idx  = lane & 15;
  const int sub  = lane >> 4;
  const int ks   = sub * 8;

  // staging coordinates
  const int ar = tid >> 2;          // 0..63
  const int ac = (tid & 3) * 8;     // 0,8,16,24
  const int br = tid >> 1;          // 0..127
  const int bc = (tid & 1) * 16;    // 0,16

  const unsigned bdst0 = lds_off32(&Bs[0][br * LDB + bc]);
  const unsigned bdst1 = lds_off32(&Bs[1][br * LDB + bc]);

  auto stageB = [&](int k0, int buf) {
    const bf16_t* g = WxT + (size_t)(n0 + br) * K + k0 + bc;
    const unsigned dst = buf ? bdst1 : bdst0;
    async_copy_b128(dst,      g);
    async_copy_b128(dst + 16, g + 8);
  };
  auto stageA = [&](int k0, int buf) {
    const float4* p = (const float4*)(x + (size_t)(r0 + ar) * K + k0 + ac);
    float4 f0 = p[0], f1 = p[1];
    v8bf pk;
    pk[0] = f2bf(f0.x); pk[1] = f2bf(f0.y); pk[2] = f2bf(f0.z); pk[3] = f2bf(f0.w);
    pk[4] = f2bf(f1.x); pk[5] = f2bf(f1.y); pk[6] = f2bf(f1.z); pk[7] = f2bf(f1.w);
    *(v8bf*)&As[buf][ar * LDA + ac] = pk;
  };

  v8f c00 = {}, c01 = {}, c10 = {}, c11 = {};

  auto compute = [&](int buf) {
    Frag16 a0, a1, b0, b1;
    const bf16_t* ap0 = &As[buf][(wm * 32 +      idx) * LDA];
    const bf16_t* ap1 = &As[buf][(wm * 32 + 16 + idx) * LDA];
    const bf16_t* bp0 = &Bs[buf][(wn * 32 +      idx) * LDB];
    const bf16_t* bp1 = &Bs[buf][(wn * 32 + 16 + idx) * LDB];
    a0.h[0] = *(const v8bf*)(ap0 + ks);  a0.h[1] = *(const v8bf*)(ap0 + 16 + ks);
    a1.h[0] = *(const v8bf*)(ap1 + ks);  a1.h[1] = *(const v8bf*)(ap1 + 16 + ks);
    b0.h[0] = *(const v8bf*)(bp0 + ks);  b0.h[1] = *(const v8bf*)(bp0 + 16 + ks);
    b1.h[0] = *(const v8bf*)(bp1 + ks);  b1.h[1] = *(const v8bf*)(bp1 + 16 + ks);
    c00 = __builtin_amdgcn_wmma_f32_16x16x32_bf16(false, a0.v, false, b0.v, (short)0, c00, false, false);
    c01 = __builtin_amdgcn_wmma_f32_16x16x32_bf16(false, a0.v, false, b1.v, (short)0, c01, false, false);
    c10 = __builtin_amdgcn_wmma_f32_16x16x32_bf16(false, a1.v, false, b0.v, (short)0, c10, false, false);
    c11 = __builtin_amdgcn_wmma_f32_16x16x32_bf16(false, a1.v, false, b1.v, (short)0, c11, false, false);
  };

  // prologue: chunks 0 and 1 in flight
  stageB(0, 0);      stageA(0, 0);
  stageB(BK, 1);     stageA(BK, 1);

  // steady state: branch-free (always wait<=2, compute, restage at distance 2)
#pragma unroll 2
  for (int i = 0; i < NCH - 2; ++i) {
    const int buf = i & 1;
    wait_async2();                  // chunk i landed (in-order completion)
    __syncthreads();                // publish chunk i
    compute(buf);
    __syncthreads();                // all reads of buf done (DScnt drained)
    stageB((i + 2) * BK, buf);      // refill the buffer chunk i vacated
    stageA((i + 2) * BK, buf);
  }

  // drain: chunk NCH-2 (buf 0), then chunk NCH-1 (buf 1); no restaging
  wait_async2();
  __syncthreads();
  compute((NCH - 2) & 1);
  wait_async0();
  __syncthreads();
  compute((NCH - 1) & 1);

  // epilogue: + bias, store f32 (C layout: vgpr r -> M = sub*8 + r, N = idx)
  const int col0 = n0 + wn * 32 + idx;
  const int col1 = col0 + 16;
  const float bv0 = bias[col0];
  const float bv1 = bias[col1];
#pragma unroll
  for (int r = 0; r < 8; ++r) {
    const int row0 = r0 + wm * 32 + sub * 8 + r;
    const int row1 = row0 + 16;
    out[(size_t)row0 * NC + col0] = c00[r] + bv0;
    out[(size_t)row0 * NC + col1] = c01[r] + bv1;
    out[(size_t)row1 * NC + col0] = c10[r] + bv0;
    out[(size_t)row1 * NC + col1] = c11[r] + bv1;
  }
}

// ------------- phase 2: persistent scan  h_t = tanh(xW_t + h @ Wh) ---------
// 64 blocks x 256 thr (8 waves). Block b owns hidden cols [16b, 16b+16);
// wave w owns batch rows [16w, 16w+16). Wh slice (16x1024 bf16 = 32KB) is
// staged once into LDS via async copies and reused for all 256 steps.

#define LDSW 1032   // padded Wh LDS row stride (halves): 2064B, 16B-aligned

__global__ __launch_bounds__(256) void rnn_scan(
    const bf16_t* __restrict__ WhT,   // (1024, 1024) [j][k] bf16
    float*        __restrict__ out,   // (128, 256, 1024) f32, in-place xW -> h
    bf16_t*       __restrict__ hb0,   // (128, 1024) bf16 ping
    bf16_t*       __restrict__ hb1,   // (128, 1024) bf16 pong
    unsigned*     __restrict__ gbar)  // grid-barrier counter (zeroed per call)
{
  const int H = 1024, T = 256, NBLK = 64;
  __shared__ __align__(16) bf16_t Ws[16 * LDSW];

  const int tid  = threadIdx.x;
  const int lane = tid & 31;
  const int wave = tid >> 5;           // batch tile 0..7
  const int idx  = lane & 15;
  const int sub  = lane >> 4;
  const int ks   = sub * 8;
  const int nc   = blockIdx.x * 16;    // hidden-col base
  const int mb   = wave * 16;          // batch-row base
  const int col  = nc + idx;

  // one-time stage of Wh slice: rows nc..nc+15, all K, async global->LDS
  {
    const int row = tid >> 4;                // 0..15
    const int seg = (tid & 15) * 128;        // byte offset within 2048B row
    const unsigned ldst = lds_off32(&Ws[row * LDSW]) + (unsigned)seg;
    const char* g = (const char*)(WhT + (size_t)(nc + row) * H) + seg;
#pragma unroll
    for (int j = 0; j < 8; ++j)
      async_copy_b128(ldst + j * 16, g + j * 16);
  }
  wait_async0();
  __syncthreads();

  const bf16_t* bp = &Ws[idx * LDSW];

  for (int t = 0; t < T; ++t) {
    const bf16_t* hcur = (t & 1) ? hb1 : hb0;
    bf16_t*       hnxt = (t & 1) ? hb0 : hb1;

    // C init = xW slice (f32, already includes bias); split accumulators to
    // halve the dependent-WMMA chain on the step critical path.
    v8f ca, cb = {};
#pragma unroll
    for (int r = 0; r < 8; ++r)
      ca[r] = out[((size_t)(mb + sub * 8 + r) * T + t) * H + col];

    const bf16_t* ar = hcur + (size_t)(mb + idx) * H;
#pragma unroll 4
    for (int k = 0; k < H; k += 64) {
      Frag16 a0, b0, a1, b1;
      a0.h[0] = *(const v8bf*)(ar + k + ks);
      a0.h[1] = *(const v8bf*)(ar + k + 16 + ks);
      b0.h[0] = *(const v8bf*)(bp + k + ks);
      b0.h[1] = *(const v8bf*)(bp + k + 16 + ks);
      a1.h[0] = *(const v8bf*)(ar + k + 32 + ks);
      a1.h[1] = *(const v8bf*)(ar + k + 48 + ks);
      b1.h[0] = *(const v8bf*)(bp + k + 32 + ks);
      b1.h[1] = *(const v8bf*)(bp + k + 48 + ks);
      ca = __builtin_amdgcn_wmma_f32_16x16x32_bf16(false, a0.v, false, b0.v, (short)0, ca, false, false);
      cb = __builtin_amdgcn_wmma_f32_16x16x32_bf16(false, a1.v, false, b1.v, (short)0, cb, false, false);
    }

#pragma unroll
    for (int r = 0; r < 8; ++r) {
      const int row = mb + sub * 8 + r;
      const float hv = tanhf(ca[r] + cb[r]);
      out[((size_t)row * T + t) * H + col] = hv;
      hnxt[(size_t)row * H + col] = f2bf(hv);
    }

    // ---- grid-wide barrier (release h_t, acquire before reading it) ----
    __builtin_amdgcn_fence(__ATOMIC_RELEASE, "agent");
    __syncthreads();
    if (tid == 0) {
      __hip_atomic_fetch_add(gbar, 1u, __ATOMIC_RELAXED, __HIP_MEMORY_SCOPE_AGENT);
      const unsigned target = (unsigned)(t + 1) * (unsigned)NBLK;
      while (__hip_atomic_load(gbar, __ATOMIC_RELAXED, __HIP_MEMORY_SCOPE_AGENT) < target)
        __builtin_amdgcn_s_sleep(2);
    }
    __syncthreads();
    __builtin_amdgcn_fence(__ATOMIC_ACQUIRE, "agent");
  }
}

// ---------------------------------------------------------------------------

extern "C" void kernel_launch(void* const* d_in, const int* in_sizes, int n_in,
                              void* d_out, int out_size, void* d_ws, size_t ws_size,
                              hipStream_t stream) {
  const float* x  = (const float*)d_in[0];
  const float* h0 = (const float*)d_in[1];
  const float* Wx = (const float*)d_in[2];
  const float* Wh = (const float*)d_in[3];
  const float* b  = (const float*)d_in[4];
  float* out = (float*)d_out;

  const int D = 1024, H = 1024, N = 128, T = 256;

  // workspace: WxT(2MB) | WhT(2MB) | h ping-pong (2 x 256KB) | barrier ctr
  char* ws = (char*)d_ws;
  bf16_t*   WxT  = (bf16_t*)(ws);
  bf16_t*   WhT  = (bf16_t*)(ws + (size_t)(2u << 20));
  bf16_t*   hb0  = (bf16_t*)(ws + (size_t)(4u << 20));
  bf16_t*   hb1  = (bf16_t*)(ws + (size_t)(4u << 20) + (size_t)(256u << 10));
  unsigned* gbar = (unsigned*)(ws + (size_t)(4u << 20) + (size_t)(512u << 10));

  zero_u32<<<1, 64, 0, stream>>>(gbar);
  cvt_transpose<<<(D * H + 255) / 256, 256, 0, stream>>>(Wx, WxT, D, H);
  cvt_transpose<<<(H * H + 255) / 256, 256, 0, stream>>>(Wh, WhT, H, H);
  cvt<<<(N * H + 255) / 256, 256, 0, stream>>>(h0, hb0, N * H);

  dim3 g1((N * T) / BM, H / BN);
  xw_gemm<<<g1, 256, 0, stream>>>(x, WxT, b, out);

  rnn_scan<<<H / 16, 256, 0, stream>>>(WhT, out, hb0, hb1, gbar);
}
